// HetEncoder_17918603559563
// MI455X (gfx1250) — matrix operand order repaired and checked by reading, not compiled, not run on hardware
//
#include <hip/hip_runtime.h>
#include <hip/hip_bf16.h>
#include <math.h>

// ---------------------------------------------------------------------------
// CDNA5 (gfx1250) heterogeneous graph-transformer encoder.
// Dense matmuls on v_wmma_f32_16x16x32_bf16 (fp32 accum); LDS tiles stored in
// fragment-friendly layouts so fragments load as ds_load_b128 pairs.
// ---------------------------------------------------------------------------

typedef __attribute__((ext_vector_type(16))) __bf16       v16bf;
typedef __attribute__((ext_vector_type(8)))  float        v8f;
typedef __attribute__((ext_vector_type(8)))  unsigned short us8;

union BFrag { v16bf v; unsigned short u[16]; us8 h[2]; };

__device__ __forceinline__ unsigned short f2bf(float f) {
  unsigned u = __float_as_uint(f);
  unsigned r = (u + 0x7fffu + ((u >> 16) & 1u)) >> 16;   // RNE truncate to bf16
  return (unsigned short)r;
}

__device__ __forceinline__ void atomicMaxF(float* addr, float val) {
  unsigned int* a = (unsigned int*)addr;
  unsigned int old = *a;
  while (true) {
    float f = __uint_as_float(old);
    if (f >= val) break;
    unsigned int prev = atomicCAS(a, old, __float_as_uint(val));
    if (prev == old) break;
    old = prev;
  }
}

// ---------------------------------------------------------------------------
// GEMM: C[M,N] = A[M,K](f32) * B[K,N](f32) + bias[N], bf16 WMMA, f32 accum.
// Block tile 64x128, BK=32, 256 threads = 8 waves; wave w owns rows
// (w>>1)*16..+15 and cols (w&1)*64..+63 (4 accumulators of 16x16).
// LDS layouts:
//   sA[row*32 + k]  (row-major)   -> A frag = 2x contiguous ds_load_b128
//   sB[col*32 + k]  (col-major)   -> B frag = 2x contiguous ds_load_b128
// N multiple of 128 (here {128,256}); K multiple of 32.
// ---------------------------------------------------------------------------
#define BM 64
#define BN 128
#define BK 32

__global__ __launch_bounds__(256)
void k_gemm_bias(const float* __restrict__ A, const float* __restrict__ B,
                 const float* __restrict__ bias, float* __restrict__ C,
                 int M, int N, int K, int lda, int ldb, int ldc)
{
  __shared__ alignas(16) unsigned short sA[BM * BK];   // [row][k]
  __shared__ alignas(16) unsigned short sB[BN * BK];   // [col][k]
  const int tid  = threadIdx.x;
  const int lane = tid & 31;
  const int wave = tid >> 5;
  const int brow = blockIdx.y * BM;
  const int bcol = blockIdx.x * BN;
  const int wrow = (wave >> 1) * 16;
  const int wcol = (wave & 1) * 64;
  const int mloc  = lane & 15;   // row within 16 (A) / col within 16 (B,C)
  const int khalf = lane >> 4;   // K-half selector

  v8f acc[4] = {v8f{}, v8f{}, v8f{}, v8f{}};

  for (int k0 = 0; k0 < K; k0 += BK) {
    // ---- Stage A tile: 64x32 = 512 float4 slots, 2 per thread ----
#pragma unroll
    for (int i = 0; i < 2; ++i) {
      int slot = i * 256 + tid;
      int r = slot >> 3, c4 = (slot & 7) << 2;
      int gr = brow + r;
      float4 v = make_float4(0.f, 0.f, 0.f, 0.f);
      if (gr < M) v = *(const float4*)&A[(size_t)gr * lda + k0 + c4];
      sA[r * 32 + c4 + 0] = f2bf(v.x);
      sA[r * 32 + c4 + 1] = f2bf(v.y);
      sA[r * 32 + c4 + 2] = f2bf(v.z);
      sA[r * 32 + c4 + 3] = f2bf(v.w);
    }
    // ---- Stage B tile: 32x128 = 1024 float4 slots, 4 per thread ----
#pragma unroll
    for (int i = 0; i < 4; ++i) {
      int slot = i * 256 + tid;
      int r = slot >> 5, c4 = (slot & 31) << 2;
      float4 v = *(const float4*)&B[(size_t)(k0 + r) * ldb + bcol + c4];
      sB[(c4 + 0) * 32 + r] = f2bf(v.x);   // column-major store
      sB[(c4 + 1) * 32 + r] = f2bf(v.y);
      sB[(c4 + 2) * 32 + r] = f2bf(v.z);
      sB[(c4 + 3) * 32 + r] = f2bf(v.w);
    }
    __syncthreads();

    // Prefetch next K tile into cache (global_prefetch_b8).
    if (k0 + BK < K) {
      int pr = brow + (tid & 63); if (pr >= M) pr = M - 1;
      __builtin_prefetch(&A[(size_t)pr * lda + k0 + BK + ((tid >> 6) << 3)], 0, 3);
      __builtin_prefetch(&B[(size_t)(k0 + BK + (tid & 31)) * ldb + bcol + ((tid >> 5) << 4)], 0, 3);
    }

    // A fragment: lanes 0-15 rows, halves {K0..7,K16..23}; lanes 16-31 {8..15,24..31}.
    BFrag a;
    a.h[0] = *(const us8*)&sA[(wrow + mloc) * 32 + khalf * 8];
    a.h[1] = *(const us8*)&sA[(wrow + mloc) * 32 + 16 + khalf * 8];

    // 4 B fragments (32x16): same K halves as A, {K0..7,K16..23} / {8..15,24..31}, from col-major sB.
#pragma unroll
    for (int t = 0; t < 4; ++t) {
      BFrag b;
      const unsigned short* pb = &sB[(wcol + t * 16 + mloc) * 32 + khalf * 8];
      b.h[0] = *(const us8*)&pb[0];
      b.h[1] = *(const us8*)&pb[16];
      acc[t] = __builtin_amdgcn_wmma_f32_16x16x32_bf16(
          false, a.v, false, b.v, (short)0, acc[t], false, false);
    }
    __syncthreads();
  }

  // Epilogue: C layout VGPR i -> row = i + 8*(lane>>4), col = lane&15.
#pragma unroll
  for (int t = 0; t < 4; ++t) {
#pragma unroll
    for (int i = 0; i < 8; ++i) {
      int gr = brow + wrow + i + 8 * khalf;
      int gc = bcol + wcol + t * 16 + mloc;
      if (gr < M) {
        float bv = bias ? bias[gc] : 0.0f;
        C[(size_t)gr * ldc + gc] = acc[t][i] + bv;
      }
    }
  }
}

// ---------------------------------------------------------------------------
// Relation transform: Y[n, h*32+e] = sum_d X[n, h*32+d] * W[h, d, e].
// One wave handles 16 nodes for head blockIdx.y. The 32x32 head matrix is
// staged once per block into LDS column-major (sW[e*32+d]) so B fragments
// load as contiguous ds_load_b128 pairs. A rows load as float4 from global.
// Grid: (ceil(N/128), NHEAD), 256 threads (8 waves x 16 nodes).
// ---------------------------------------------------------------------------
__global__ __launch_bounds__(256)
void k_rel(const float* __restrict__ X, const float* __restrict__ W,
           float* __restrict__ Y, int Ntype)
{
  __shared__ alignas(16) unsigned short sW[32 * 32];   // [e][d] col-major vs W
  const int tid  = threadIdx.x;
  const int lane = tid & 31;
  const int wave = tid >> 5;
  const int n0 = (blockIdx.x * 8 + wave) * 16;
  const int h  = blockIdx.y;
  const int mloc = lane & 15, khalf = lane >> 4;

  // Stage W[h]: 1024 floats = 256 float4 slots, 1 per thread.
  {
    const float* w = W + (size_t)h * 1024;           // [d][e] row-major
    int d = tid >> 3, e0 = (tid & 7) << 2;
    float4 v = *(const float4*)&w[d * 32 + e0];
    sW[(e0 + 0) * 32 + d] = f2bf(v.x);
    sW[(e0 + 1) * 32 + d] = f2bf(v.y);
    sW[(e0 + 2) * 32 + d] = f2bf(v.z);
    sW[(e0 + 3) * 32 + d] = f2bf(v.w);
  }
  __syncthreads();

  int node = n0 + mloc; if (node >= Ntype) node = Ntype - 1;   // clamp loads
  const float* xr = X + (size_t)node * 256 + h * 32;
  BFrag a;
  {
    float4 lo = *(const float4*)&xr[khalf * 8];
    float4 lo2 = *(const float4*)&xr[khalf * 8 + 4];
    float4 hi = *(const float4*)&xr[16 + khalf * 8];
    float4 hi2 = *(const float4*)&xr[16 + khalf * 8 + 4];
    a.u[0] = f2bf(lo.x);  a.u[1] = f2bf(lo.y);  a.u[2] = f2bf(lo.z);  a.u[3] = f2bf(lo.w);
    a.u[4] = f2bf(lo2.x); a.u[5] = f2bf(lo2.y); a.u[6] = f2bf(lo2.z); a.u[7] = f2bf(lo2.w);
    a.u[8]  = f2bf(hi.x);  a.u[9]  = f2bf(hi.y);  a.u[10] = f2bf(hi.z);  a.u[11] = f2bf(hi.w);
    a.u[12] = f2bf(hi2.x); a.u[13] = f2bf(hi2.y); a.u[14] = f2bf(hi2.z); a.u[15] = f2bf(hi2.w);
  }
  BFrag b0, b1;
  b0.h[0] = *(const us8*)&sW[mloc * 32 + khalf * 8];
  b0.h[1] = *(const us8*)&sW[mloc * 32 + 16 + khalf * 8];
  b1.h[0] = *(const us8*)&sW[(16 + mloc) * 32 + khalf * 8];
  b1.h[1] = *(const us8*)&sW[(16 + mloc) * 32 + 16 + khalf * 8];

  v8f c0 = {}, c1 = {};
  c0 = __builtin_amdgcn_wmma_f32_16x16x32_bf16(false, a.v, false, b0.v, (short)0, c0, false, false);
  c1 = __builtin_amdgcn_wmma_f32_16x16x32_bf16(false, a.v, false, b1.v, (short)0, c1, false, false);
#pragma unroll
  for (int i = 0; i < 8; ++i) {
    int r = n0 + i + 8 * khalf;
    if (r < Ntype) {
      Y[(size_t)r * 256 + h * 32 + mloc]      = c0[i];
      Y[(size_t)r * 256 + h * 32 + 16 + mloc] = c1[i];
    }
  }
}

// ---------------------------------------------------------------------------
// Edge kernels
// ---------------------------------------------------------------------------
__global__ void k_edge_alpha(const int* __restrict__ src, const int* __restrict__ dst,
                             const float* __restrict__ Q, const float* __restrict__ KR,
                             const float* __restrict__ prio, float* __restrict__ alpha,
                             float* __restrict__ amax, int offd, int E, float scale)
{
  long long idx = (long long)blockIdx.x * blockDim.x + threadIdx.x;
  if (idx >= (long long)E * 8) return;
  int e = (int)(idx >> 3), h = (int)(idx & 7);
  int s = src[e], d = dst[e];
  const float4* q = (const float4*)(Q  + (size_t)d * 256 + h * 32);
  const float4* k = (const float4*)(KR + (size_t)s * 256 + h * 32);
  float acc = 0.0f;
#pragma unroll
  for (int j = 0; j < 8; ++j) {
    float4 a = q[j], b = k[j];
    acc += a.x * b.x + a.y * b.y + a.z * b.z + a.w * b.w;
  }
  acc *= prio[h] * scale;
  alpha[(size_t)e * 8 + h] = acc;
  atomicMaxF(&amax[(size_t)(offd + d) * 8 + h], acc);
}

__global__ void k_amax_fix(float* __restrict__ amax, long long n)
{
  long long i = (long long)blockIdx.x * blockDim.x + threadIdx.x;
  if (i < n) { float v = amax[i]; if (!__builtin_isfinite(v)) amax[i] = 0.0f; }
}

__global__ void k_exp_den(const int* __restrict__ dst, const float* __restrict__ alpha,
                          float* __restrict__ exb, const float* __restrict__ amax,
                          float* __restrict__ den, int offd, int E)
{
  long long idx = (long long)blockIdx.x * blockDim.x + threadIdx.x;
  if (idx >= (long long)E * 8) return;
  int e = (int)(idx >> 3), h = (int)(idx & 7);
  int d = dst[e];
  float ex = expf(alpha[(size_t)e * 8 + h] - amax[(size_t)(offd + d) * 8 + h]);
  exb[(size_t)e * 8 + h] = ex;
  atomicAdd(&den[(size_t)(offd + d) * 8 + h], ex);
}

__global__ void k_scatter(const int* __restrict__ src, const int* __restrict__ dst,
                          const float* __restrict__ VR, const float* __restrict__ exb,
                          const float* __restrict__ den, float* __restrict__ agg,
                          int offd, int E)
{
  long long idx = (long long)blockIdx.x * 256 + threadIdx.x;   // E blocks x 256
  int e = (int)(idx >> 8), c = (int)(idx & 255), h = c >> 5;
  int s = src[e], d = dst[e];
  float co = exb[(size_t)e * 8 + h] /
             fmaxf(den[(size_t)(offd + d) * 8 + h], 1e-16f);
  atomicAdd(&agg[(size_t)(offd + d) * 256 + c], VR[(size_t)s * 256 + c] * co);
}

// ---------------------------------------------------------------------------
// Elementwise / normalization kernels
// ---------------------------------------------------------------------------
__global__ void k_fill(float* __restrict__ p, float v, long long n)
{
  long long i = (long long)blockIdx.x * blockDim.x + threadIdx.x;
  if (i < n) p[i] = v;
}

__global__ void k_gelu(float* __restrict__ y, const float* __restrict__ x, long long n)
{
  long long i = (long long)blockIdx.x * blockDim.x + threadIdx.x;
  if (i < n) { float v = x[i]; y[i] = 0.5f * v * (1.0f + erff(v * 0.70710678118f)); }
}

__global__ void k_combine(float* __restrict__ h, const float* __restrict__ o,
                          const float* __restrict__ skip, long long n)
{
  long long i = (long long)blockIdx.x * blockDim.x + threadIdx.x;
  if (i < n) {
    float g = 1.0f / (1.0f + expf(-skip[0]));
    // hn = g*o + (1-g)*h ; h = hn + h  ->  h = g*o + (2-g)*h
    h[i] = g * o[i] + (2.0f - g) * h[i];
  }
}

// Wave-per-row LayerNorm over D=256, optional ReLU, in place.
__global__ __launch_bounds__(256)
void k_ln256(float* __restrict__ x, const float* __restrict__ g,
             const float* __restrict__ b, int rows, int relu)
{
  int lane = threadIdx.x & 31, wave = threadIdx.x >> 5;
  int row = blockIdx.x * 8 + wave;
  if (row >= rows) return;                       // wave-uniform
  float* xr = x + (size_t)row * 256;
  float v[8], s = 0.0f, s2 = 0.0f;
#pragma unroll
  for (int i = 0; i < 8; ++i) { v[i] = xr[lane + 32 * i]; s += v[i]; s2 += v[i] * v[i]; }
#pragma unroll
  for (int m = 16; m >= 1; m >>= 1) { s += __shfl_xor(s, m, 32); s2 += __shfl_xor(s2, m, 32); }
  float mean = s * (1.0f / 256.0f);
  float var  = s2 * (1.0f / 256.0f) - mean * mean;
  float inv  = rsqrtf(var + 1e-5f);
#pragma unroll
  for (int i = 0; i < 8; ++i) {
    float y = (v[i] - mean) * inv * g[lane + 32 * i] + b[lane + 32 * i];
    if (relu) y = fmaxf(y, 0.0f);
    xr[lane + 32 * i] = y;
  }
}

// Wave-per-row LayerNorm (D=128) fused with L2 row normalization -> out.
__global__ __launch_bounds__(256)
void k_lnl2(const float* __restrict__ Z, const float* __restrict__ g,
            const float* __restrict__ b, float* __restrict__ out, int rows)
{
  int lane = threadIdx.x & 31, wave = threadIdx.x >> 5;
  int row = blockIdx.x * 8 + wave;
  if (row >= rows) return;
  const float* z = Z + (size_t)row * 128;
  float v[4], s = 0.0f, s2 = 0.0f;
#pragma unroll
  for (int i = 0; i < 4; ++i) { v[i] = z[lane + 32 * i]; s += v[i]; s2 += v[i] * v[i]; }
#pragma unroll
  for (int m = 16; m >= 1; m >>= 1) { s += __shfl_xor(s, m, 32); s2 += __shfl_xor(s2, m, 32); }
  float mean = s * (1.0f / 128.0f);
  float var  = s2 * (1.0f / 128.0f) - mean * mean;
  float inv  = rsqrtf(var + 1e-5f);
  float y[4], nrm = 0.0f;
#pragma unroll
  for (int i = 0; i < 4; ++i) {
    y[i] = (v[i] - mean) * inv * g[lane + 32 * i] + b[lane + 32 * i];
    nrm += y[i] * y[i];
  }
#pragma unroll
  for (int m = 16; m >= 1; m >>= 1) nrm += __shfl_xor(nrm, m, 32);
  float n = fmaxf(sqrtf(nrm), 1e-12f);
#pragma unroll
  for (int i = 0; i < 4; ++i) out[(size_t)row * 128 + lane + 32 * i] = y[i] / n;
}

// ---------------------------------------------------------------------------
// Host orchestration
// ---------------------------------------------------------------------------
extern "C" void kernel_launch(void* const* d_in, const int* in_sizes, int n_in,
                              void* d_out, int out_size, void* d_ws, size_t ws_size,
                              hipStream_t stream)
{
  (void)in_sizes; (void)n_in; (void)out_size; (void)ws_size;

  // Node types in NT order: user, event, space, tag.
  static const int NN[4]    = {100000, 50000, 5000, 2000};
  static const int OFF[4]   = {0, 100000, 150000, 155000};
  static const int TOT      = 157000;
  static const int INDIM[4] = {256, 384, 128, 64};
  static const int SIDX[4]  = {3, 0, 1, 2};   // NT idx -> sorted (event,space,tag,user)

  // Edge types in ET order; rel params are flattened in sorted-name order.
  static const int ESRC[8] = {0, 1, 1, 2, 1, 3, 0, 3};
  static const int EDST[8] = {1, 0, 2, 1, 3, 1, 3, 0};
  static const int EC[8]   = {400000, 400000, 50000, 50000, 100000, 100000, 100000, 100000};
  static const int RSORT[8]= {0, 4, 2, 6, 1, 5, 3, 7};
  static const int EOFF[8] = {0, 400000, 800000, 850000, 900000, 1000000, 1100000, 1200000};
  static const long long ETOT = 1300000;

  // Param leaf indices (JAX pytree: sorted dict keys).
  const int PIN = 12;        // in.{event,space,tag,user}.{W,b,beta,g}
  const int PL0 = 28;        // layers[0]
  const int PLS = 60;        // per-layer leaf count
  const int POUT = 148;      // out.{W,b,beta,g}

  auto f = [&](int i) { return (const float*)d_in[i]; };

  // Workspace carve-out (floats).
  float* ws = (float*)d_ws;
  size_t p = 0;
  auto alloc = [&](size_t n) { float* r = ws + p; p += n; return r; };
  float* h    = alloc((size_t)TOT * 256);
  float* Kb   = alloc((size_t)TOT * 256);
  float* Qb   = alloc((size_t)TOT * 256);
  float* Vb   = alloc((size_t)TOT * 256);
  float* agg  = alloc((size_t)TOT * 256);
  float* REL  = alloc((size_t)100000 * 256);   // per-relation k_rel / v_rel scratch
  float* alp  = alloc((size_t)ETOT * 8);
  float* exb  = alloc((size_t)ETOT * 8);
  float* amax = alloc((size_t)TOT * 8);
  float* den  = alloc((size_t)TOT * 8);
  float* gbuf = Kb;   // reuse: K dead after attention logits
  float* obuf = Qb;   // reuse: Q dead after attention logits
  float* zbuf = Vb;   // reuse: V dead after scatter of last layer

  const float scale = 0.17677669529663687f;   // 1/sqrt(32)

  auto gemm = [&](const float* A, const float* B, const float* bias, float* C,
                  int M, int N, int K, int lda, int ldb, int ldc) {
    dim3 grid(N / 128, (M + BM - 1) / BM);
    k_gemm_bias<<<grid, 256, 0, stream>>>(A, B, bias, C, M, N, K, lda, ldb, ldc);
  };
  auto fill = [&](float* ptr, float v, long long n) {
    k_fill<<<(unsigned)((n + 255) / 256), 256, 0, stream>>>(ptr, v, n);
  };

  // ---- Input projections + LN + ReLU -> h ----
  for (int t = 0; t < 4; ++t) {
    int ib = PIN + SIDX[t] * 4;
    gemm((const float*)d_in[t], f(ib + 0), f(ib + 1),
         h + (size_t)OFF[t] * 256, NN[t], 256, INDIM[t], INDIM[t], 256, 256);
    k_ln256<<<(NN[t] + 7) / 8, 256, 0, stream>>>(h + (size_t)OFF[t] * 256,
                                                 f(ib + 3), f(ib + 2), NN[t], 1);
  }

  // ---- Layers ----
  for (int L = 0; L < 2; ++L) {
    int lb = PL0 + L * PLS;

    // K/Q/V projections per type.
    for (int t = 0; t < 4; ++t) {
      int kb = lb + SIDX[t] * 6;     // Wk,Wq,Wv,bk,bq,bv
      const float* ht = h + (size_t)OFF[t] * 256;
      gemm(ht, f(kb + 0), f(kb + 3), Kb + (size_t)OFF[t] * 256, NN[t], 256, 256, 256, 256, 256);
      gemm(ht, f(kb + 1), f(kb + 4), Qb + (size_t)OFF[t] * 256, NN[t], 256, 256, 256, 256, 256);
      gemm(ht, f(kb + 2), f(kb + 5), Vb + (size_t)OFF[t] * 256, NN[t], 256, 256, 256, 256, 256);
    }

    fill(amax, -INFINITY, (long long)TOT * 8);
    fill(den, 0.0f, (long long)TOT * 8);
    fill(agg, 0.0f, (long long)TOT * 256);

    // Pass A: per relation, k_rel = K x a[h], alpha + atomic segment max.
    for (int r = 0; r < 8; ++r) {
      int s = ESRC[r], d = EDST[r], E = EC[r];
      const int* esrc = (const int*)d_in[4 + r];
      const int* edst = esrc + E;
      int rb = lb + 32 + RSORT[r] * 3;   // a,m,p
      k_rel<<<dim3((NN[s] + 127) / 128, 8), 256, 0, stream>>>(
          Kb + (size_t)OFF[s] * 256, f(rb + 0), REL, NN[s]);
      long long n = (long long)E * 8;
      k_edge_alpha<<<(unsigned)((n + 255) / 256), 256, 0, stream>>>(
          esrc, edst, Qb + (size_t)OFF[d] * 256, REL, f(rb + 2),
          alp + (size_t)EOFF[r] * 8, amax, OFF[d], E, scale);
    }
    k_amax_fix<<<(TOT * 8 + 255) / 256, 256, 0, stream>>>(amax, (long long)TOT * 8);

    // Pass B: exp + segment denominators.
    for (int r = 0; r < 8; ++r) {
      int d = EDST[r], E = EC[r];
      const int* esrc = (const int*)d_in[4 + r];
      const int* edst = esrc + E;
      long long n = (long long)E * 8;
      k_exp_den<<<(unsigned)((n + 255) / 256), 256, 0, stream>>>(
          edst, alp + (size_t)EOFF[r] * 8, exb + (size_t)EOFF[r] * 8,
          amax, den, OFF[d], E);
    }

    // Pass C: v_rel = V x m[h], then weighted scatter-add into agg.
    for (int r = 0; r < 8; ++r) {
      int s = ESRC[r], d = EDST[r], E = EC[r];
      const int* esrc = (const int*)d_in[4 + r];
      const int* edst = esrc + E;
      int rb = lb + 32 + RSORT[r] * 3;
      k_rel<<<dim3((NN[s] + 127) / 128, 8), 256, 0, stream>>>(
          Vb + (size_t)OFF[s] * 256, f(rb + 1), REL, NN[s]);
      k_scatter<<<(unsigned)E, 256, 0, stream>>>(
          esrc, edst, REL, exb + (size_t)EOFF[r] * 8, den, agg, OFF[d], E);
    }

    // GELU -> output projection -> gated skip + residual.
    k_gelu<<<(unsigned)(((long long)TOT * 256 + 255) / 256), 256, 0, stream>>>(
        gbuf, agg, (long long)TOT * 256);
    for (int t = 0; t < 4; ++t) {
      int ob = lb + 24 + SIDX[t] * 2;    // W,b
      gemm(gbuf + (size_t)OFF[t] * 256, f(ob + 0), f(ob + 1),
           obuf + (size_t)OFF[t] * 256, NN[t], 256, 256, 256, 256, 256);
      k_combine<<<(unsigned)(((long long)NN[t] * 256 + 255) / 256), 256, 0, stream>>>(
          h + (size_t)OFF[t] * 256, obuf + (size_t)OFF[t] * 256,
          f(lb + 56 + SIDX[t]), (long long)NN[t] * 256);
    }
  }

  // ---- Final projection + LN + L2 normalize -> d_out ----
  for (int t = 0; t < 4; ++t) {
    gemm(h + (size_t)OFF[t] * 256, f(POUT + 0), f(POUT + 1),
         zbuf + (size_t)OFF[t] * 128, NN[t], 128, 256, 256, 128, 128);
    k_lnl2<<<(NN[t] + 7) / 8, 256, 0, stream>>>(
        zbuf + (size_t)OFF[t] * 128, f(POUT + 3), f(POUT + 2),
        (float*)d_out + (size_t)OFF[t] * 128, NN[t]);
  }
}
